// Decoder_2594160247151
// MI455X (gfx1250) — matrix operand order repaired and checked by reading, not compiled, not run on hardware
//
#include <hip/hip_runtime.h>

typedef _Float16 f16;
typedef __attribute__((ext_vector_type(8)))  _Float16 v8h;
typedef __attribute__((ext_vector_type(16))) _Float16 v16h;
typedef __attribute__((ext_vector_type(8)))  float    v8f;

#define Q0 50000
#define Q1 10000

// ---------------- small prep kernels ----------------

__global__ void zero_f32(float* __restrict__ p, int n) {
    int i = blockIdx.x * blockDim.x + threadIdx.x;
    if (i < n) p[i] = 0.0f;
}

// one thread per (row, feature): atomic segment sums + counts
__global__ void seg_accum(const float* __restrict__ emb, const int* __restrict__ z,
                          float* __restrict__ s, float* __restrict__ c, int n) {
    int i = blockIdx.x * blockDim.x + threadIdx.x;
    if (i >= n * 16) return;
    int row = i >> 4, f = i & 15;
    int g = z[row];
    atomicAdd(&s[(size_t)g * 16 + f], emb[i]);
    if (f == 0) atomicAdd(&c[g], 1.0f);
}

// divide_no_nan -> f16 group-mean table
__global__ void seg_mean(const float* __restrict__ s, const float* __restrict__ c,
                         f16* __restrict__ B, int q) {
    int i = blockIdx.x * blockDim.x + threadIdx.x;
    if (i >= q * 16) return;
    float cc = c[i >> 4];
    B[i] = (cc > 0.0f) ? (f16)(s[i] / cc) : (f16)0.0f;
}

// Pack W1 (64x128) and W2 (128x64) into WMMA B-fragment layout:
//   fragment index = ((ntile*KC + kchunk)*32 + lane)*16 + i
//   half i of lane L maps to K = kchunk*32 + (L>=16 ? 8 : 0) + (i&7) + (i>=8 ? 16 : 0), N = ntile*16 + (L&15)
__global__ void pack_weights(const float* __restrict__ W1, const float* __restrict__ W2,
                             f16* __restrict__ w1p, f16* __restrict__ w2p) {
    for (int idx = blockIdx.x * blockDim.x + threadIdx.x; idx < 8192;
         idx += blockDim.x * gridDim.x) {
        int i    = idx & 15;
        int lane = (idx >> 4) & 31;
        int kg   = (lane >> 4) << 4;          // 0 or 16
        // W1: 8 ntiles x 2 kchunks
        {
            int kc = (idx >> 9) & 1, nt = idx >> 10;
            int k = kc * 32 + (kg >> 1) + (i & 7) + ((i >> 3) << 4);
            int nn = nt * 16 + (lane & 15);
            w1p[idx] = (f16)W1[k * 128 + nn];
        }
        // W2: 4 ntiles x 4 kchunks
        {
            int kc = (idx >> 9) & 3, nt = idx >> 11;
            int k = kc * 32 + (kg >> 1) + (i & 7) + ((i >> 3) << 4);
            int nn = nt * 16 + (lane & 15);
            w2p[idx] = (f16)W2[k * 64 + nn];
        }
    }
}

// ---------------- fused gather + 3-layer MLP ----------------

// A-fragment (16x32 f16): lane<16 holds row M=lane, K = {0..7, 16..23}+0;
// lane>=16 holds row M=lane-16, K = {0..7, 16..23}+8. Two 16B LDS loads.
__device__ __forceinline__ v16h ldfragA(const f16* p) {
    v8h lo = *(const v8h*)p;
    v8h hi = *(const v8h*)(p + 16);
    return __builtin_shufflevector(lo, hi, 0, 1, 2, 3, 4, 5, 6, 7,
                                           8, 9, 10, 11, 12, 13, 14, 15);
}

__device__ __forceinline__ v8f wmma16(v16h a, v16h b, v8f c) {
    return __builtin_amdgcn_wmma_f32_16x16x32_f16(false, a, false, b,
                                                  (short)0, c, false, false);
}

__launch_bounds__(128)
__global__ void fused_mlp(const float* __restrict__ X,
                          const int* __restrict__ z0, const int* __restrict__ z1,
                          const f16* __restrict__ B0, const f16* __restrict__ B1,
                          const f16* __restrict__ w1p, const f16* __restrict__ w2p,
                          const float* __restrict__ b1, const float* __restrict__ b2,
                          const float* __restrict__ Wout, const float* __restrict__ bout,
                          float* __restrict__ out, int n, int ntiles) {
    __shared__ __align__(16) f16  w1s[8192];       // 16 KB
    __shared__ __align__(16) f16  w2s[8192];       // 16 KB
    __shared__ float b1s[128];
    __shared__ float b2s[64];
    __shared__ float wos[64];
    __shared__ __align__(16) f16  h0s[4][16 * 64]; //  8 KB
    __shared__ __align__(16) f16  h1s[4][16 * 128];// 16 KB

    const int tid = threadIdx.x;
    for (int i = tid; i < 8192; i += 128) { w1s[i] = w1p[i]; w2s[i] = w2p[i]; }
    if (tid < 128) b1s[tid] = b1[tid];
    if (tid < 64)  { b2s[tid] = b2[tid]; wos[tid] = Wout[tid]; }
    __syncthreads();

    const int wave = tid >> 5;
    const int lane = tid & 31;
    const int tile = blockIdx.x * 4 + wave;
    if (tile >= ntiles) return;                 // wave-uniform exit

    f16* h0 = h0s[wave];
    f16* h1 = h1s[wave];
    const int row0  = tile * 16;
    const int lrow  = lane & 15;                // col within N-tile / A row
    const int koff  = (lane >> 4) << 3;         // A-layout K shift: 0 or 8
    const int mbase = (lane >> 4) << 3;         // C-layout row base: 0 or 8

    // ---- build h0 = [X | zb0 | zb1] as f16 in LDS (zero-padded tail) ----
    for (int rr = 0; rr < 16; ++rr) {
        int r = row0 + rr;
        float xv = 0.0f;
        if (r < n) xv = X[(size_t)r * 32 + lane];
        h0[rr * 64 + lane] = (f16)xv;
        if (lane < 16) {
            f16 e0 = (f16)0.0f, e1 = (f16)0.0f;
            if (r < n) {
                e0 = B0[(size_t)z0[r] * 16 + lane];
                e1 = B1[(size_t)z1[r] * 16 + lane];
            }
            h0[rr * 64 + 32 + lane] = e0;
            h0[rr * 64 + 48 + lane] = e1;
        }
    }
    asm volatile("s_wait_dscnt 0" ::: "memory");   // in-wave LDS handoff

    // ---- layer 1: [16x64] @ [64x128] + b1, relu -> h1 ----
    v16h a0 = ldfragA(h0 + lrow * 64 + 0 * 32 + koff);
    v16h a1 = ldfragA(h0 + lrow * 64 + 1 * 32 + koff);
    for (int nt = 0; nt < 8; ++nt) {
        float bv = b1s[nt * 16 + lrow];
        v8f c = {bv, bv, bv, bv, bv, bv, bv, bv};
        v16h bf0 = *(const v16h*)(w1s + (((nt * 2 + 0) * 32 + lane) << 4));
        v16h bf1 = *(const v16h*)(w1s + (((nt * 2 + 1) * 32 + lane) << 4));
        c = wmma16(a0, bf0, c);
        c = wmma16(a1, bf1, c);
        #pragma unroll
        for (int v = 0; v < 8; ++v) {
            float rv = fmaxf(c[v], 0.0f);
            h1[(mbase + v) * 128 + nt * 16 + lrow] = (f16)rv;
        }
    }
    asm volatile("s_wait_dscnt 0" ::: "memory");

    // ---- layer 2: [16x128] @ [128x64] + b2, relu, fused @Wout ----
    v16h a2[4];
    #pragma unroll
    for (int kc = 0; kc < 4; ++kc)
        a2[kc] = ldfragA(h1 + lrow * 128 + kc * 32 + koff);

    float racc[8];
    #pragma unroll
    for (int v = 0; v < 8; ++v) racc[v] = 0.0f;

    for (int nt = 0; nt < 4; ++nt) {
        float bv = b2s[nt * 16 + lrow];
        v8f c = {bv, bv, bv, bv, bv, bv, bv, bv};
        #pragma unroll
        for (int kc = 0; kc < 4; ++kc) {
            v16h bf = *(const v16h*)(w2s + (((nt * 4 + kc) * 32 + lane) << 4));
            c = wmma16(a2[kc], bf, c);
        }
        float wv = wos[nt * 16 + lrow];
        #pragma unroll
        for (int v = 0; v < 8; ++v)
            racc[v] += fmaxf(c[v], 0.0f) * wv;
    }

    // reduce each row across its 16 column-lanes (lanes 0..15 / 16..31 groups)
    #pragma unroll
    for (int v = 0; v < 8; ++v) {
        #pragma unroll
        for (int off = 1; off < 16; off <<= 1)
            racc[v] += __shfl_xor(racc[v], off, 32);
    }
    if (lrow == 0) {
        float bo = bout[0];
        #pragma unroll
        for (int v = 0; v < 8; ++v) {
            int r = row0 + mbase + v;
            if (r < n) out[r] = racc[v] + bo;
        }
    }
}

// ---------------- host launcher ----------------

extern "C" void kernel_launch(void* const* d_in, const int* in_sizes, int n_in,
                              void* d_out, int out_size, void* d_ws, size_t ws_size,
                              hipStream_t stream) {
    const float* X    = (const float*)d_in[0];
    const int*   z0   = (const int*)d_in[1];
    const int*   z1   = (const int*)d_in[2];
    const float* emb0 = (const float*)d_in[3];
    const float* emb1 = (const float*)d_in[4];
    const float* W1   = (const float*)d_in[5];
    const float* b1   = (const float*)d_in[6];
    const float* W2   = (const float*)d_in[7];
    const float* b2   = (const float*)d_in[8];
    const float* Wout = (const float*)d_in[9];
    const float* bout = (const float*)d_in[10];
    float* out = (float*)d_out;
    const int n = in_sizes[1];                 // N (length of z0)

    // workspace layout: [s0 | c0 | s1 | c1] f32 (zeroed), then f16 tables
    float* s0 = (float*)d_ws;
    float* c0 = s0 + (size_t)Q0 * 16;
    float* s1 = c0 + Q0;
    float* c1 = s1 + (size_t)Q1 * 16;
    f16* B0  = (f16*)(c1 + Q1);
    f16* B1  = B0 + (size_t)Q0 * 16;
    f16* w1p = B1 + (size_t)Q1 * 16;
    f16* w2p = w1p + 8192;

    const int zn = Q0 * 17 + Q1 * 17;
    zero_f32<<<(zn + 255) / 256, 256, 0, stream>>>(s0, zn);
    seg_accum<<<(n * 16 + 255) / 256, 256, 0, stream>>>(emb0, z0, s0, c0, n);
    seg_accum<<<(n * 16 + 255) / 256, 256, 0, stream>>>(emb1, z1, s1, c1, n);
    seg_mean<<<(Q0 * 16 + 255) / 256, 256, 0, stream>>>(s0, c0, B0, Q0);
    seg_mean<<<(Q1 * 16 + 255) / 256, 256, 0, stream>>>(s1, c1, B1, Q1);
    pack_weights<<<32, 256, 0, stream>>>(W1, W2, w1p, w2p);

    const int ntiles = (n + 15) / 16;
    fused_mlp<<<(ntiles + 3) / 4, 128, 0, stream>>>(X, z0, z1, B0, B1, w1p, w2p,
                                                    b1, b2, Wout, bout, out, n, ntiles);
}
